// LiteformerFastAttention_12171937317201
// MI455X (gfx1250) — matrix operand order repaired and checked by reading, not compiled, not run on hardware
//
#include <hip/hip_runtime.h>
#include <hip/hip_bf16.h>
#include <math.h>

typedef __attribute__((ext_vector_type(16))) _Float16 v16h;
typedef __attribute__((ext_vector_type(8)))  float    v8f;
typedef __attribute__((ext_vector_type(4)))  unsigned v4u;

#define H_    16
#define N_    8192
#define C_    64
#define M_    256
#define K_    64
#define RPB   64            // rows per workgroup block
#define NBLK  (N_ / RPB)    // 128
#define BIAS  65.0f
#define NBIAS 532480.0f     // N_ * (nbits+1)

// ---------------------------------------------------------------------------
// WMMA fragment helpers for v_wmma_f32_16x16x32_f16 (CDNA5 ISA 7.12.2 layouts)
// ---------------------------------------------------------------------------

// A operand (16x32 f16): lane m=L%16, g=L/16; pair p -> K = 8g+2p (p<4) else 16+8g+2(p-4)
__device__ inline v16h load_a_frag(const _Float16* lds, int row_base, int ldk,
                                   int k_base, int lane) {
  int m = lane & 15, g = lane >> 4;
  const _Float16* rp = lds + (size_t)(row_base + m) * ldk + k_base;
  union { v16h v; unsigned u[8]; } r;
#pragma unroll
  for (int p = 0; p < 8; ++p) {
    int kk = (p < 4) ? (8 * g + 2 * p) : (16 + 8 * g + 2 * (p - 4));
    r.u[p] = *(const unsigned*)(rp + kk);
  }
  return r.v;
}

// A operand built from f32 LDS with per-row scale (qk row normalization fused)
__device__ inline v16h load_a_frag_f32_scaled(const float* lds, const float* rn,
                                              int row_base, int ldk, int k_base, int lane) {
  int m = lane & 15, g = lane >> 4;
  float s = rn[row_base + m];
  const float* rp = lds + (size_t)(row_base + m) * ldk + k_base;
  union { v16h v; _Float16 h[16]; } r;
#pragma unroll
  for (int p = 0; p < 8; ++p) {
    int kk = (p < 4) ? (8 * g + 2 * p) : (16 + 8 * g + 2 * (p - 4));
    r.h[2 * p]     = (_Float16)(rp[kk] * s);
    r.h[2 * p + 1] = (_Float16)(rp[kk + 1] * s);
  }
  return r.v;
}

// B operand (32x16 f16) from "BT" layout BT[n][k] (K contiguous per lane):
// lane n=L%16, g=L/16; pair p -> K = 8g+2p (p<4) else 16+8g+2(p-4)
__device__ inline v16h load_bt_frag(const _Float16* lds, int n_base, int ldk,
                                    int k_base, int lane) {
  int n = lane & 15, g = lane >> 4;
  const _Float16* cp = lds + (size_t)(n_base + n) * ldk + k_base;
  union { v16h v; unsigned u[8]; } r;
#pragma unroll
  for (int p = 0; p < 8; ++p) r.u[p] = *(const unsigned*)(cp + ((p < 4) ? (8 * g + 2 * p) : (16 + 8 * g + 2 * (p - 4))));
  return r.v;
}

// D (16x16 f32): VGPR r, lanes 0-15 -> (M=r, N=lane); lanes 16-31 -> (M=r+8, N=lane-16)
__device__ inline void store_d_rowmajor_f16(_Float16* lds, int ldn, int n_base,
                                            const v8f& d, int lane) {
  int n = lane & 15, g = lane >> 4;
#pragma unroll
  for (int r = 0; r < 8; ++r)
    lds[(size_t)(r + 8 * g) * ldn + n_base + n] = (_Float16)d[r];
}

// D stored transposed: T[n][row] with r contiguous -> one aligned 16B LDS store
__device__ inline void store_d_transposed_f16(_Float16* lds, int n_base, int ldr,
                                              int row_base, const v8f& d, int lane) {
  int n = lane & 15, g = lane >> 4;
  union { v4u v; _Float16 h[8]; } t;
#pragma unroll
  for (int r = 0; r < 8; ++r) t.h[r] = (_Float16)d[r];
  *(v4u*)(lds + (size_t)(n_base + n) * ldr + row_base + 8 * g) = t.v;
}

// ---------------------------------------------------------------------------
// Shared prolog: stage anchor/W_hash/qk to LDS, row norms, GEMM1 (sim)
// ---------------------------------------------------------------------------
__device__ inline void stage_and_gemm1(
    const float* __restrict__ qk_blk, const float* __restrict__ anchor_h,
    const float* __restrict__ whash_h,
    _Float16* anchorH, _Float16* whashT, float* stage, float* rnorm,
    _Float16* simW, v16h* a0_out, v16h* a1_out, int tid, int lane, int wav) {
  // qk block [64][64] f32 -> LDS (coalesced float4)
  const float4* q4 = (const float4*)qk_blk;
  float4* s4 = (float4*)stage;
  for (int i = tid; i < RPB * C_ / 4; i += 128) s4[i] = q4[i];

  // anchor [256][64] -> f16 LDS same layout (== BT layout for GEMM1)
  const float4* a4 = (const float4*)anchor_h;
  for (int i = tid; i < M_ * C_ / 4; i += 128) {
    float4 t = a4[i];
    _Float16* dst = anchorH + i * 4;
    dst[0] = (_Float16)t.x; dst[1] = (_Float16)t.y;
    dst[2] = (_Float16)t.z; dst[3] = (_Float16)t.w;
  }
  // W_hash [m][64] -> whashT [bit][m] f16 (transpose during staging)
  const float4* w4 = (const float4*)whash_h;
  for (int i = tid; i < M_ * K_ / 4; i += 128) {
    float4 t = w4[i];
    int m = (i * 4) >> 6, k = (i * 4) & 63;
    whashT[(size_t)(k + 0) * M_ + m] = (_Float16)t.x;
    whashT[(size_t)(k + 1) * M_ + m] = (_Float16)t.y;
    whashT[(size_t)(k + 2) * M_ + m] = (_Float16)t.z;
    whashT[(size_t)(k + 3) * M_ + m] = (_Float16)t.w;
  }
  __syncthreads();

  // per-row 1/||qk||
  if (tid < RPB) {
    const float4* r4 = (const float4*)(stage + tid * C_);
    float acc = 0.f;
#pragma unroll
    for (int i = 0; i < 16; ++i) {
      float4 t = r4[i];
      acc += t.x * t.x + t.y * t.y + t.z * t.z + t.w * t.w;
    }
    rnorm[tid] = 1.0f / sqrtf(acc);
  }
  __syncthreads();

  // normalized qk rows -> A fragments (one 16-row tile per wave)
  v16h a0 = load_a_frag_f32_scaled(stage, rnorm, 16 * wav, C_, 0, lane);
  v16h a1 = load_a_frag_f32_scaled(stage, rnorm, 16 * wav, C_, 32, lane);
  *a0_out = a0; *a1_out = a1;

  // GEMM1: sim[16][256] = qkn @ anchor^T (16 n-tiles x 2 k-steps)
  for (int nt = 0; nt < 16; ++nt) {
    v8f c = {};
    v16h b0 = load_bt_frag(anchorH, nt * 16, C_, 0, lane);
    v16h b1 = load_bt_frag(anchorH, nt * 16, C_, 32, lane);
    c = __builtin_amdgcn_wmma_f32_16x16x32_f16(false, a0, false, b0, (short)0, c, false, false);
    c = __builtin_amdgcn_wmma_f32_16x16x32_f16(false, a1, false, b1, (short)0, c, false, false);
    store_d_rowmajor_f16(simW, M_, nt * 16, c, lane);
  }
}

// ---------------------------------------------------------------------------
// Kernel 0: zero ctx + ksum workspace
// ---------------------------------------------------------------------------
__global__ void lfa_init(float* p, int n) {
  int i = blockIdx.x * 256 + threadIdx.x;
  if (i < n) p[i] = 0.f;
}

// ---------------------------------------------------------------------------
// Kernel 1: per block, phi = tanh(sim @ W_hash); accumulate ctx = phi^T v, ksum
// ---------------------------------------------------------------------------
__global__ __launch_bounds__(128) void lfa_pass_ctx(
    const float* __restrict__ qk, const float* __restrict__ v,
    const float* __restrict__ anchor, const float* __restrict__ whash,
    float* __restrict__ ctx_g, float* __restrict__ ksum_g) {
  __shared__ __align__(16) _Float16 anchorH[M_ * C_];
  __shared__ __align__(16) _Float16 whashT[K_ * M_];
  __shared__ __align__(16) _Float16 simH[4 * 16 * M_];
  __shared__ __align__(16) float    stage[RPB * C_];
  __shared__ float    rnorm[RPB];
  __shared__ __align__(16) _Float16 phiT[K_ * 80];   // phiT[bit][row], ld=80
  __shared__ __align__(16) _Float16 vT[C_ * 80];     // vT[e][row],   ld=80
  __shared__ float    ksumL[K_];

  const int tid = threadIdx.x, lane = tid & 31, wav = tid >> 5;
  const int blk = blockIdx.x, bh = blockIdx.y;
  const int h = bh & (H_ - 1);
  const size_t base_g = ((size_t)bh * N_ + (size_t)blk * RPB) * (size_t)C_;

  if (tid < K_) ksumL[tid] = 0.f;

  v16h a0, a1;
  stage_and_gemm1(qk + base_g, anchor + (size_t)h * M_ * C_,
                  whash + (size_t)h * M_ * K_, anchorH, whashT, stage, rnorm,
                  simH + (size_t)wav * 16 * M_, &a0, &a1, tid, lane, wav);

  // v block -> vT f16 (transpose during staging)
  const float* vblk = v + base_g;
  for (int i = tid; i < RPB * C_; i += 128) {
    int r = i >> 6, e = i & 63;
    vT[(size_t)e * 80 + r] = (_Float16)vblk[i];
  }

  const _Float16* simW = simH + (size_t)wav * 16 * M_;
  // GEMM2 + tanh -> phiT + ksum partials
  for (int bt = 0; bt < 4; ++bt) {
    v8f c = {};
#pragma unroll
    for (int s = 0; s < 8; ++s) {
      v16h a = load_a_frag(simW, 0, M_, 32 * s, lane);
      v16h b = load_bt_frag(whashT, bt * 16, M_, 32 * s, lane);
      c = __builtin_amdgcn_wmma_f32_16x16x32_f16(false, a, false, b, (short)0, c, false, false);
    }
    float loc = 0.f;
#pragma unroll
    for (int r = 0; r < 8; ++r) { c[r] = tanhf(c[r]); loc += c[r]; }
    atomicAdd(&ksumL[bt * 16 + (lane & 15)], loc);
    store_d_transposed_f16(phiT, bt * 16, 80, 16 * wav, c, lane);
  }
  __syncthreads();

  // ctx[k][e] += phiT(k,:) @ v(:,e); wave w owns k-tile w
  float* ctxbh = ctx_g + (size_t)bh * K_ * C_;
  for (int et = 0; et < 4; ++et) {
    v8f c = {};
#pragma unroll
    for (int s = 0; s < 2; ++s) {
      v16h a = load_a_frag(phiT, 16 * wav, 80, 32 * s, lane);
      v16h b = load_bt_frag(vT, 16 * et, 80, 32 * s, lane);
      c = __builtin_amdgcn_wmma_f32_16x16x32_f16(false, a, false, b, (short)0, c, false, false);
    }
    int n = lane & 15, g = lane >> 4;
#pragma unroll
    for (int r = 0; r < 8; ++r)
      atomicAdd(ctxbh + (size_t)(16 * wav + r + 8 * g) * C_ + et * 16 + n, c[r]);
  }
  if (tid < K_) atomicAdd(ksum_g + (size_t)bh * K_ + tid, ksumL[tid]);
}

// ---------------------------------------------------------------------------
// Kernel 2: recompute phi; out = (phi @ ctx + v*bias) / (phi . ksum + N*bias)
// ---------------------------------------------------------------------------
__global__ __launch_bounds__(128) void lfa_pass_out(
    const float* __restrict__ qk, const float* __restrict__ v,
    const float* __restrict__ anchor, const float* __restrict__ whash,
    const float* __restrict__ ctx_g, const float* __restrict__ ksum_g,
    float* __restrict__ out) {
  __shared__ __align__(16) _Float16 anchorH[M_ * C_];
  __shared__ __align__(16) _Float16 whashT[K_ * M_];
  __shared__ __align__(16) _Float16 simH[4 * 16 * M_];
  __shared__ __align__(16) float    stage[RPB * C_];   // qk, later v
  __shared__ float    rnorm[RPB];
  __shared__ __align__(16) _Float16 phiA[4 * 16 * K_]; // per wave, A layout source
  __shared__ __align__(16) _Float16 ctxT[C_ * K_];     // ctxT[e][k]
  __shared__ float    ksumL[K_];
  __shared__ float    dinv[RPB];

  const int tid = threadIdx.x, lane = tid & 31, wav = tid >> 5;
  const int blk = blockIdx.x, bh = blockIdx.y;
  const int h = bh & (H_ - 1);
  const size_t base_g = ((size_t)bh * N_ + (size_t)blk * RPB) * (size_t)C_;

  v16h a0, a1;
  stage_and_gemm1(qk + base_g, anchor + (size_t)h * M_ * C_,
                  whash + (size_t)h * M_ * K_, anchorH, whashT, stage, rnorm,
                  simH + (size_t)wav * 16 * M_, &a0, &a1, tid, lane, wav);

  // ctx -> ctxT f16, ksum -> LDS
  const float* ctxbh = ctx_g + (size_t)bh * K_ * C_;
  for (int i = tid; i < K_ * C_; i += 128) {
    int k = i >> 6, e = i & 63;
    ctxT[(size_t)e * K_ + k] = (_Float16)ctxbh[i];
  }
  if (tid < K_) ksumL[tid] = ksum_g[(size_t)bh * K_ + tid];

  const _Float16* simW = simH + (size_t)wav * 16 * M_;
  _Float16* phiW = phiA + (size_t)wav * 16 * K_;
  // GEMM2 + tanh -> phiA (row-major [16][64], == A-operand source)
  for (int bt = 0; bt < 4; ++bt) {
    v8f c = {};
#pragma unroll
    for (int s = 0; s < 8; ++s) {
      v16h a = load_a_frag(simW, 0, M_, 32 * s, lane);
      v16h b = load_bt_frag(whashT, bt * 16, M_, 32 * s, lane);
      c = __builtin_amdgcn_wmma_f32_16x16x32_f16(false, a, false, b, (short)0, c, false, false);
    }
#pragma unroll
    for (int r = 0; r < 8; ++r) c[r] = tanhf(c[r]);
    store_d_rowmajor_f16(phiW, K_, bt * 16, c, lane);
  }
  __syncthreads();  // phiA, ctxT, ksumL visible; all waves past stage reads

  // denominators: D_inv[row] = 1 / (phi[row] . ksum + N*bias)
  if (lane < 16) {
    const _Float16* pr = phiW + (size_t)lane * K_;
    float acc = 0.f;
#pragma unroll
    for (int k2 = 0; k2 < K_; ++k2) acc += (float)pr[k2] * ksumL[k2];
    dinv[wav * 16 + lane] = 1.0f / (acc + NBIAS);
  }
  __syncthreads();

  // v block -> stage (reuse qk staging buffer)
  const float4* v4 = (const float4*)(v + base_g);
  float4* s4 = (float4*)stage;
  for (int i = tid; i < RPB * C_ / 4; i += 128) s4[i] = v4[i];
  __syncthreads();

  // out GEMM: out[16][64] = phi[16][64] @ ctx[64][64], then bias/denominator
  v16h ap0 = load_a_frag(phiW, 0, K_, 0, lane);
  v16h ap1 = load_a_frag(phiW, 0, K_, 32, lane);
  float* outbh = out + base_g;
  for (int et = 0; et < 4; ++et) {
    v8f c = {};
    v16h b0 = load_bt_frag(ctxT, 16 * et, K_, 0, lane);
    v16h b1 = load_bt_frag(ctxT, 16 * et, K_, 32, lane);
    c = __builtin_amdgcn_wmma_f32_16x16x32_f16(false, ap0, false, b0, (short)0, c, false, false);
    c = __builtin_amdgcn_wmma_f32_16x16x32_f16(false, ap1, false, b1, (short)0, c, false, false);
    int n = lane & 15, g = lane >> 4;
#pragma unroll
    for (int r = 0; r < 8; ++r) {
      int row = 16 * wav + r + 8 * g;
      float o = (c[r] + stage[(size_t)row * C_ + et * 16 + n] * BIAS) * dinv[row];
      outbh[(size_t)row * C_ + et * 16 + n] = o;
    }
  }
}

// ---------------------------------------------------------------------------
extern "C" void kernel_launch(void* const* d_in, const int* in_sizes, int n_in,
                              void* d_out, int out_size, void* d_ws, size_t ws_size,
                              hipStream_t stream) {
  const float* qk     = (const float*)d_in[0];
  const float* v      = (const float*)d_in[1];
  const float* anchor = (const float*)d_in[2];
  const float* whash  = (const float*)d_in[3];
  float* out  = (float*)d_out;
  float* ctx  = (float*)d_ws;                 // [32][64][64] f32
  float* ksum = ctx + 32 * K_ * C_;           // [32][64]     f32

  int nz = 32 * K_ * C_ + 32 * K_;            // 133120 floats (~532 KB)
  lfa_init<<<(nz + 255) / 256, 256, 0, stream>>>(ctx, nz);

  dim3 grid(NBLK, 32);
  lfa_pass_ctx<<<grid, 128, 0, stream>>>(qk, v, anchor, whash, ctx, ksum);
  lfa_pass_out<<<grid, 128, 0, stream>>>(qk, v, anchor, whash, ctx, ksum, out);
}
